// InnerFaceShiftTriple_84146999263923
// MI455X (gfx1250) — matrix operand order repaired and checked by reading, not compiled, not run on hardware
//
#include <hip/hip_runtime.h>

typedef __attribute__((ext_vector_type(16))) _Float16 v16h;
typedef __attribute__((ext_vector_type(8)))  _Float16 v8h;
typedef __attribute__((ext_vector_type(8)))  float    v8f;

#define NB   8
#define CH   256
#define HW   4096
#define MM   8192
#define KK   256
#define EPSF 1e-8f
#define NEGF -1000000000.0f
#define BSTRIDE 272   // LDS B-tile row stride in halves (256 + 16 pad)
#define MTILE 32      // m-columns per block iteration (2 x 16-wide WMMA tiles)

#if __has_builtin(__builtin_amdgcn_tensor_load_to_lds) && \
    __has_builtin(__builtin_amdgcn_s_wait_tensorcnt)
#define USE_TDM 1
#endif

#ifdef USE_TDM
typedef unsigned int v4u __attribute__((ext_vector_type(4)));
typedef int          v4i __attribute__((ext_vector_type(4)));
typedef int          v8i __attribute__((ext_vector_type(8)));

// LDS byte offset of a __shared__ object (addrspacecast -> 32-bit offset)
__device__ __forceinline__ unsigned lds_offset(const void* p)
{
    return (unsigned)(size_t)(const __attribute__((address_space(3))) void*)p;
}

// TDM: load a 32-row x 512B tile (row stride 512B) into LDS, padding each
// 512B row with 8 DWORDs so LDS rows land on BSTRIDE=272 halves.
__device__ __forceinline__ void tdm_load_tile(const _Float16* gsrc, unsigned lds_addr)
{
    unsigned long long ga = (unsigned long long)(size_t)gsrc;
    v4u g0;
    g0[0] = 1u;                                   // count=1 (valid user D#)
    g0[1] = lds_addr;                             // lds_addr (bytes)
    g0[2] = (unsigned)(ga & 0xFFFFFFFFu);         // global_addr[31:0]
    g0[3] = (unsigned)((ga >> 32) & 0x01FFFFFFu)  // global_addr[56:32]
          | 0x80000000u;                          // type=2 ("image")
    v8i g1;
    g1[0] = (int)0x0F910000u;     // data_size=1(2B) | pad_enable | interval=6 | amount=7
    g1[1] = (int)(256u << 16);    // tensor_dim0[15:0]=256
    g1[2] = (int)(32u  << 16);    // tensor_dim0 hi=0 | tensor_dim1[15:0]=32
    g1[3] = (int)(256u << 16);    // tensor_dim1 hi=0 | tile_dim0=256
    g1[4] = 32;                   // tile_dim1=32 | tile_dim2=0
    g1[5] = 256;                  // tensor_dim0_stride[31:0]=256 elements
    g1[6] = 0;
    g1[7] = 0;
    v4i z4 = {0, 0, 0, 0};
#if defined(__clang_major__) && __clang_major__ >= 23
    v8i z8 = {0, 0, 0, 0, 0, 0, 0, 0};
    __builtin_amdgcn_tensor_load_to_lds(g0, g1, z4, z4, z8, 0);
#else
    __builtin_amdgcn_tensor_load_to_lds(g0, g1, z4, z4, 0);
#endif
}
#endif // USE_TDM

// ---------------------------------------------------------------------------
// Kernel 1: per-position L2 norms (latter / former / flip_feat) + valid bias
// ---------------------------------------------------------------------------
__global__ void k_norms(const float* __restrict__ inp,   // B,512,HW
                        const float* __restrict__ flip,  // B,256,HW
                        const int*   __restrict__ mask,  // B,HW
                        float* __restrict__ normLat,     // B*HW
                        float* __restrict__ normCand,    // B*MM
                        float* __restrict__ bias)        // B*MM
{
    int gid = blockIdx.x * blockDim.x + threadIdx.x;     // NB*3*HW threads
    int b = gid / (3 * HW);
    int p = gid % (3 * HW);
    const float* src;
    if (p < HW)        src = inp  + ((size_t)b * 2 * CH + CH) * HW + p;   // latter
    else if (p < 2*HW) src = inp  + (size_t)b * 2 * CH * HW + (p - HW);   // former
    else               src = flip + (size_t)b * CH * HW + (p - 2 * HW);   // flip_feat
    float s = 0.f;
    #pragma unroll 4
    for (int c = 0; c < CH; ++c) {
        float v = src[(size_t)c * HW];
        s += v * v;
    }
    float nv = sqrtf(s) + EPSF;
    if (p < HW) {
        normLat[b * HW + p] = nv;
    } else {
        int m = p - HW;                  // 0..MM
        normCand[b * MM + m] = nv;
        int pix;
        if (m < HW) pix = m;
        else { int mm = m - HW; pix = (mm & ~63) + (63 - (mm & 63)); }    // W-flip
        bias[b * MM + m] = (mask[b * HW + pix] == 0) ? 0.f : NEGF;
    }
}

// ---------------------------------------------------------------------------
// Kernel 2: normalize + f32->f16 + transpose to k-contiguous Ah[n][k], Bh[m][k]
// ---------------------------------------------------------------------------
__global__ void k_convert(const float* __restrict__ inp,
                          const float* __restrict__ flip,
                          const float* __restrict__ normLat,
                          const float* __restrict__ normCand,
                          _Float16* __restrict__ Ah,     // B,HW,KK
                          _Float16* __restrict__ Bh)     // B,MM,KK
{
    __shared__ _Float16 tile[32][33];
    int b  = blockIdx.z;
    int ct = blockIdx.x;          // k tile 0..7
    int pt = blockIdx.y;          // pos tile 0..383
    int tx = threadIdx.x;         // 0..31
    int ty = threadIdx.y;         // 0..7
    int p0 = pt * 32;
    const float* src; const float* nrm; _Float16* dst;
    if (p0 < HW) {                // latter -> Ah
        src = inp + ((size_t)b * 2 * CH + CH) * HW + p0;
        nrm = normLat + b * HW + p0;
        dst = Ah + ((size_t)b * HW + p0) * KK;
    } else {
        int m0 = p0 - HW;
        nrm = normCand + b * MM + m0;
        dst = Bh + ((size_t)b * MM + m0) * KK;
        if (m0 < HW) src = inp  + (size_t)b * 2 * CH * HW + m0;           // former
        else         src = flip + (size_t)b * CH * HW + (m0 - HW);        // flip
    }
    int c0 = ct * 32;
    float nv = nrm[tx];
    #pragma unroll
    for (int i = 0; i < 4; ++i) {
        int c = ty + i * 8;
        float v = src[(size_t)(c0 + c) * HW + tx];
        tile[c][tx] = (_Float16)(v / nv);
    }
    __syncthreads();
    #pragma unroll
    for (int i = 0; i < 4; ++i) {
        int pr = ty + i * 8;                          // pos within tile
        dst[(size_t)pr * KK + c0 + tx] = tile[tx][pr];
    }
}

// ---------------------------------------------------------------------------
// Kernel 3: fused WMMA GEMM (sim = Ah * Bh^T) + bias + running argmax over m
// block = 256 threads (8 waves); wave owns a 16-row n-strip, A kept in VGPRs.
// B tiles (32 m-cols, 16KB) double-buffered in LDS:
//   - TDM path: wave 0 issues tensor_load_to_lds, syncs via TENSORcnt+barrier
//   - fallback: cooperative register staging
// Two independent accumulator chains -> 16 WMMA per barrier.
// ---------------------------------------------------------------------------
__global__ void __launch_bounds__(256) k_gemm_argmax(
    const _Float16* __restrict__ Ah,
    const _Float16* __restrict__ Bh,
    const float* __restrict__ bias,
    int* __restrict__ outIdx)
{
    __shared__ _Float16 Bt[2][MTILE * BSTRIDE];   // 2 * 32 * 272 * 2B = 34816 B
    int b    = blockIdx.y;
    int tid  = threadIdx.x;
    int wave = tid >> 5;
    int lane = tid & 31;
    int col  = lane & 15;         // N (m_local) / M-row for A loads
    int hi   = lane >> 4;         // upper half of wave
    int nbase = blockIdx.x * 128 + wave * 16;

    const _Float16* Abase = Ah + (size_t)b * HW * KK;
    const _Float16* Bbase = Bh + (size_t)b * MM * KK;
    const float*    biasB = bias + (size_t)b * MM;

    // A fragments for all 8 k-steps, resident in registers.
    v16h Areg[8];
    {
        const _Float16* arow = Abase + (size_t)(nbase + col) * KK;
        int klo = hi * 16;
        #pragma unroll
        for (int ks = 0; ks < 8; ++ks) {
            v8h lo  = *(const v8h*)(arow + ks * 32 + klo);
            v8h hi8 = *(const v8h*)(arow + ks * 32 + klo + 8);
            v16h a;
            #pragma unroll
            for (int j = 0; j < 8; ++j) { a[j] = lo[j]; a[j + 8] = hi8[j]; }
            Areg[ks] = a;
        }
    }

    float best[8];
    int   bidx[8];
    #pragma unroll
    for (int r = 0; r < 8; ++r) { best[r] = -3.0e38f; bidx[r] = 0; }

    const int NIT = MM / MTILE;          // 256 iterations

#ifdef USE_TDM
    unsigned ldsBuf0 = lds_offset(&Bt[0][0]);
    unsigned ldsBuf1 = lds_offset(&Bt[1][0]);
    bool issuer = (tid < 32);            // wave 0 drives the TDM
    if (issuer) tdm_load_tile(Bbase, ldsBuf0);          // tile 0 -> buf0
#else
    // cooperative B tile staging: 32 m-cols x 256 K halves = 16KB, 64B/thread
    int mloc = tid >> 4;                 // 0..15
    int kof  = (tid & 15) * 16;          // halves
    const _Float16* bsrc0 = Bbase + (size_t)mloc * KK + kof;          // rows 0..15
    const _Float16* bsrc1 = Bbase + (size_t)(mloc + 16) * KK + kof;   // rows 16..31
    int ldoff0 = mloc * BSTRIDE + kof;
    int ldoff1 = (mloc + 16) * BSTRIDE + kof;
    v16h s0 = *(const v16h*)bsrc0;
    v16h s1 = *(const v16h*)bsrc1;
    *(v16h*)&Bt[0][ldoff0] = s0;
    *(v16h*)&Bt[0][ldoff1] = s1;
    s0 = *(const v16h*)(bsrc0 + (size_t)MTILE * KK);
    s1 = *(const v16h*)(bsrc1 + (size_t)MTILE * KK);
#endif

    for (int it = 0; it < NIT; ++it) {
#ifdef USE_TDM
        if (issuer) __builtin_amdgcn_s_wait_tensorcnt(0);  // buf[it&1] landed
        __syncthreads();                 // visible to all; prior readers fenced
        if (issuer && (it + 1 < NIT))    // next tile into the other buffer
            tdm_load_tile(Bbase + (size_t)(it + 1) * MTILE * KK,
                          (it & 1) ? ldsBuf0 : ldsBuf1);
        asm volatile("" ::: "memory");   // keep LDS loads ordered/alive
#else
        __syncthreads();   // buf[it&1] visible; prior readers of buf[(it+1)&1] done
        if (it + 1 < NIT) {
            int nb = (it + 1) & 1;
            *(v16h*)&Bt[nb][ldoff0] = s0;
            *(v16h*)&Bt[nb][ldoff1] = s1;
            if (it + 2 < NIT) {
                s0 = *(const v16h*)(bsrc0 + (size_t)(it + 2) * MTILE * KK);
                s1 = *(const v16h*)(bsrc1 + (size_t)(it + 2) * MTILE * KK);
            }
        }
#endif

        const _Float16* brow0 = &Bt[it & 1][0] + col * BSTRIDE + hi * 16;
        const _Float16* brow1 = brow0 + 16 * BSTRIDE;
        v8f acc0 = {}, acc1 = {};
        #pragma unroll
        for (int ks = 0; ks < 8; ++ks) {
            v16h bf0 = *(const v16h*)(brow0 + ks * 32);
            v16h bf1 = *(const v16h*)(brow1 + ks * 32);
            acc0 = __builtin_amdgcn_wmma_f32_16x16x32_f16(
                       false, Areg[ks], false, bf0, (short)0, acc0, false, false);
            acc1 = __builtin_amdgcn_wmma_f32_16x16x32_f16(
                       false, Areg[ks], false, bf1, (short)0, acc1, false, false);
        }

        int mbase = it * MTILE;
        float bb0 = biasB[mbase + col];
        float bb1 = biasB[mbase + 16 + col];
        #pragma unroll
        for (int r = 0; r < 8; ++r) {
            float v0 = acc0[r] + bb0;    // m = mbase + col (lower index first)
            if (v0 > best[r]) { best[r] = v0; bidx[r] = mbase + col; }
            float v1 = acc1[r] + bb1;    // m = mbase + 16 + col
            if (v1 > best[r]) { best[r] = v1; bidx[r] = mbase + 16 + col; }
        }
    }

    // reduce (max, lowest-index tie) across the 16 lanes of each half-wave
    #pragma unroll
    for (int r = 0; r < 8; ++r) {
        float v = best[r]; int iv = bidx[r];
        #pragma unroll
        for (int off = 1; off < 16; off <<= 1) {
            float ov = __shfl_xor(v, off, 32);
            int   oi = __shfl_xor(iv, off, 32);
            if (ov > v || (ov == v && oi < iv)) { v = ov; iv = oi; }
        }
        if (col == 0) outIdx[b * HW + nbase + hi * 8 + r] = iv;
    }
}

// ---------------------------------------------------------------------------
// Passthrough copies + masked gather
// ---------------------------------------------------------------------------
__global__ void k_copy_input(const float4* __restrict__ inp, float4* __restrict__ out)
{
    size_t i = (size_t)blockIdx.x * blockDim.x + threadIdx.x;   // NB*2*CH*HW/4
    size_t per = (size_t)2 * CH * HW / 4;
    size_t b = i / per, r = i % per;
    out[b * ((size_t)3 * CH * HW / 4) + r] = inp[i];
}

__global__ void k_copy_inner(const float4* __restrict__ inp, float4* __restrict__ out)
{
    size_t i = (size_t)blockIdx.x * blockDim.x + threadIdx.x;   // NB*CH*HW/4
    size_t per = (size_t)CH * HW / 4;
    size_t b = i / per, r = i % per;
    out[i] = inp[b * (size_t)2 * CH * HW / 4 + per + r];
}

__global__ void k_gather(const float* __restrict__ inp,
                         const float* __restrict__ flip,
                         const int* __restrict__ mask,
                         const int* __restrict__ outIdx,
                         float* __restrict__ out)
{
    size_t i = (size_t)blockIdx.x * blockDim.x + threadIdx.x;   // NB*CH*HW
    int n = (int)(i & (HW - 1));
    int c = (int)((i >> 12) & (CH - 1));
    int b = (int)(i >> 20);
    float v = 0.f;
    if (mask[b * HW + n] > 0) {
        int j = outIdx[b * HW + n];
        v = (j < HW) ? inp[((size_t)b * 2 * CH + c) * HW + j]
                     : flip[((size_t)b * CH + c) * HW + (j - HW)];
    }
    out[((size_t)b * 3 * CH + 2 * CH + c) * HW + n] = v;
}

// ---------------------------------------------------------------------------
extern "C" void kernel_launch(void* const* d_in, const int* in_sizes, int n_in,
                              void* d_out, int out_size, void* d_ws, size_t ws_size,
                              hipStream_t stream)
{
    const float* inp  = (const float*)d_in[0];
    const int*   mask = (const int*)d_in[1];
    const float* flip = (const float*)d_in[2];
    float* out = (float*)d_out;

    char* ws = (char*)d_ws;
    _Float16* Ah = (_Float16*)ws;                                 // 16 MB
    _Float16* Bh = (_Float16*)(ws + (size_t)NB * HW * KK * 2);    // 32 MB
    size_t off = (size_t)NB * HW * KK * 2 + (size_t)NB * MM * KK * 2;
    float* normLat  = (float*)(ws + off); off += (size_t)NB * HW * 4;
    float* normCand = (float*)(ws + off); off += (size_t)NB * MM * 4;
    float* biasArr  = (float*)(ws + off); off += (size_t)NB * MM * 4;
    int*   outIdx   = (int*)(ws + off);

    k_norms<<<NB * 3 * HW / 256, 256, 0, stream>>>(inp, flip, mask,
                                                   normLat, normCand, biasArr);

    dim3 cg(8, 384, NB), cb(32, 8);
    k_convert<<<cg, cb, 0, stream>>>(inp, flip, normLat, normCand, Ah, Bh);

    dim3 gg(HW / 128, NB);
    k_gemm_argmax<<<gg, 256, 0, stream>>>(Ah, Bh, biasArr, outIdx);

    k_copy_input<<<(NB * 2 * CH * HW / 4) / 256, 256, 0, stream>>>(
        (const float4*)inp, (float4*)out);
    k_copy_inner<<<(NB * CH * HW / 4) / 256, 256, 0, stream>>>(
        (const float4*)inp, (float4*)(out + (size_t)NB * 3 * CH * HW));
    k_gather<<<(NB * CH * HW) / 256, 256, 0, stream>>>(
        inp, flip, mask, outIdx, out);
}
